// GIN_Net_81243601371608
// MI455X (gfx1250) — compile-verified, matrix operand order checked by reading, not run on hardware
//
#include <hip/hip_runtime.h>
#include <hip/hip_bf16.h>

#define N_NODES   50000
#define N_EDGES   800000
#define DIM       128
#define OUT_CH    2
#define NUM_GRAPHS 128
#define BN_EPS    1e-5f

typedef __attribute__((ext_vector_type(16))) __bf16 v16bf;
typedef __attribute__((ext_vector_type(8)))  float  v8f;

// ---------------------------------------------------------------------------
// Vectorized copy: out = in (float4 granularity)
// ---------------------------------------------------------------------------
__global__ __launch_bounds__(256) void k_copy_f4(const float4* __restrict__ in,
                                                 float4* __restrict__ out, int n4) {
    int i = blockIdx.x * 256 + threadIdx.x;
    if (i < n4) out[i] = in[i];
}

// ---------------------------------------------------------------------------
// Zero kernel
// ---------------------------------------------------------------------------
__global__ __launch_bounds__(256) void k_zero_f(float* __restrict__ p, int n) {
    int i = blockIdx.x * 256 + threadIdx.x;
    if (i < n) p[i] = 0.0f;
}

// ---------------------------------------------------------------------------
// Edge scatter-add: agg[dst] += X[src].  One wave (32 lanes) per edge,
// each lane moves 4 contiguous floats: coalesced float4 gather +
// 4x global_atomic_add_f32 scatter.
// ---------------------------------------------------------------------------
__global__ __launch_bounds__(256) void k_scatter_add(const float* __restrict__ X,
                                                     const int* __restrict__ src,
                                                     const int* __restrict__ dst,
                                                     float* __restrict__ agg,
                                                     int n_edges) {
    int tid = blockIdx.x * 256 + threadIdx.x;
    int e = tid >> 5;
    if (e >= n_edges) return;
    int c4 = (tid & 31) * 4;
    int s = src[e], d = dst[e];
    const float4 v = *(const float4*)(X + (size_t)s * DIM + c4);
    float* p = agg + (size_t)d * DIM + c4;
    atomicAdd(p + 0, v.x);
    atomicAdd(p + 1, v.y);
    atomicAdd(p + 2, v.z);
    atomicAdd(p + 3, v.w);
}

// ---------------------------------------------------------------------------
// Global-add-pool: G[batch[node]] += H[node].  One wave per node.
// ---------------------------------------------------------------------------
__global__ __launch_bounds__(256) void k_pool_add(const float* __restrict__ H,
                                                  const int* __restrict__ batch,
                                                  float* __restrict__ G,
                                                  int n_nodes) {
    int tid = blockIdx.x * 256 + threadIdx.x;
    int node = tid >> 5;
    if (node >= n_nodes) return;
    int c4 = (tid & 31) * 4;
    int b = batch[node];
    const float4 v = *(const float4*)(H + (size_t)node * DIM + c4);
    float* p = G + (size_t)b * DIM + c4;
    atomicAdd(p + 0, v.x);
    atomicAdd(p + 1, v.y);
    atomicAdd(p + 2, v.z);
    atomicAdd(p + 3, v.w);
}

// ---------------------------------------------------------------------------
// Fused GEMM + epilogue:  out = f( A[n_rows x 128] @ W[128 x 128] + bias )
//   mode 0: f = ReLU(BN(.))   (BN folded into per-column scale/shift)
//   mode 1: f = ReLU(.)
//   mode 2: f = identity
// 256 threads = 8 waves; W staged transposed as bf16 in LDS (32 KB);
// each wave computes one 16-row strip via 8 n-tiles x 4 chained
// v_wmma_f32_16x16x32_bf16 (K = 128).
// ---------------------------------------------------------------------------
__global__ __launch_bounds__(256) void k_gemm128(const float* __restrict__ A,
                                                 const float* __restrict__ W,   // [k][n] row-major
                                                 const float* __restrict__ bias,
                                                 const float* __restrict__ bn_g,
                                                 const float* __restrict__ bn_b,
                                                 const float* __restrict__ bn_m,
                                                 const float* __restrict__ bn_v,
                                                 float* __restrict__ out,
                                                 int n_rows, int mode) {
    __shared__ __bf16 Wt[DIM * DIM];   // transposed: Wt[n][k], 32 KB

    const int t = threadIdx.x;
    // Cooperative staging of W -> LDS (transposed, converted to bf16)
    #pragma unroll
    for (int i = 0; i < (DIM * DIM) / 256; ++i) {
        int idx = t + i * 256;
        int n = idx & (DIM - 1);
        int k = idx >> 7;
        Wt[n * DIM + k] = (__bf16)W[k * DIM + n];
    }
    __syncthreads();

    const int wave = t >> 5;
    const int lane = t & 31;
    const int m16  = lane & 15;   // A row / D column within tile
    const int kh   = lane >> 4;   // K-half selector (ISA 16-bit A/B layout)

    const int strip = blockIdx.x * 8 + wave;
    if (strip * 16 >= n_rows) return;
    const int row0 = strip * 16;

    // Load this wave's 16x128 A panel as 4 K-chunks of bf16 fragments.
    // Lane layout (16-bit A 16x32): lanes 0-15 hold M=0..15; elem i<8 ->
    // K = kh*8+i, elem i>=8 -> K = 16 + kh*8 + (i-8).
    v16bf a[4];
    const float* arow = A + (size_t)(row0 + m16) * DIM;
    #pragma unroll
    for (int c = 0; c < 4; ++c) {
        const int kc = c * 32;
        #pragma unroll
        for (int i = 0; i < 8; ++i)
            a[c][i] = (__bf16)arow[kc + kh * 8 + i];
        #pragma unroll
        for (int i = 0; i < 8; ++i)
            a[c][8 + i] = (__bf16)arow[kc + 16 + kh * 8 + i];
    }

    #pragma unroll 1
    for (int nt = 0; nt < 8; ++nt) {
        const int col = nt * 16 + m16;
        v8f acc = {};
        #pragma unroll
        for (int c = 0; c < 4; ++c) {
            const int kc = c * 32;
            // B fragment: contiguous K runs from transposed LDS (16B aligned)
            const __bf16* wp = &Wt[col * DIM + kc + kh * 8];
            v16bf b;
            #pragma unroll
            for (int i = 0; i < 8; ++i) b[i] = wp[i];
            #pragma unroll
            for (int i = 0; i < 8; ++i) b[8 + i] = wp[16 + i];
            acc = __builtin_amdgcn_wmma_f32_16x16x32_bf16(
                false, a[c], false, b, (short)0, acc, false, false);
        }

        // Folded epilogue (per-column affine + optional ReLU)
        float s = 1.0f;
        float sh = bias[col];
        if (mode == 0) {
            s  = bn_g[col] * rsqrtf(bn_v[col] + BN_EPS);
            sh = bn_b[col] + (bias[col] - bn_m[col]) * s;
        }
        // D layout: VGPR v -> row v + 8*kh; lane%16 -> column
        float* op = out + (size_t)row0 * DIM + col;
        #pragma unroll
        for (int v = 0; v < 8; ++v) {
            float y = acc[v] * s + sh;
            if (mode != 2) y = fmaxf(y, 0.0f);
            op[(size_t)(v + 8 * kh) * DIM] = y;
        }
    }
}

// ---------------------------------------------------------------------------
// Final tiny linear: out[128 x 2] = G1[128 x 128] @ W[128 x 2] + b
// One block, one thread per output element.
// ---------------------------------------------------------------------------
__global__ __launch_bounds__(256) void k_final_lin(const float* __restrict__ G1,
                                                   const float* __restrict__ W,
                                                   const float* __restrict__ b,
                                                   float* __restrict__ out) {
    int t = threadIdx.x;
    int row = t >> 1;
    int col = t & 1;
    float s = b[col];
    const float* g = G1 + row * DIM;
    #pragma unroll 4
    for (int k = 0; k < DIM; ++k) s += g[k] * W[k * OUT_CH + col];
    out[row * OUT_CH + col] = s;
}

// ---------------------------------------------------------------------------
extern "C" void kernel_launch(void* const* d_in, const int* in_sizes, int n_in,
                              void* d_out, int out_size, void* d_ws, size_t ws_size,
                              hipStream_t stream) {
    const float* x      = (const float*)d_in[0];
    const int*   eidx   = (const int*)d_in[1];
    const int*   batch  = (const int*)d_in[2];
    const float* w1_1   = (const float*)d_in[3];
    const float* b1_1   = (const float*)d_in[4];
    const float* bn1_g  = (const float*)d_in[5];
    const float* bn1_b  = (const float*)d_in[6];
    const float* bn1_m  = (const float*)d_in[7];
    const float* bn1_v  = (const float*)d_in[8];
    const float* w1_2   = (const float*)d_in[9];
    const float* b1_2   = (const float*)d_in[10];
    const float* w2_1   = (const float*)d_in[11];
    const float* b2_1   = (const float*)d_in[12];
    const float* bn2_g  = (const float*)d_in[13];
    const float* bn2_b  = (const float*)d_in[14];
    const float* bn2_m  = (const float*)d_in[15];
    const float* bn2_v  = (const float*)d_in[16];
    const float* w2_2   = (const float*)d_in[17];
    const float* b2_2   = (const float*)d_in[18];
    const float* lin1_w = (const float*)d_in[19];
    const float* lin1_b = (const float*)d_in[20];
    const float* lin2_w = (const float*)d_in[21];
    const float* lin2_b = (const float*)d_in[22];

    const int* src = eidx;            // edge_index[0]
    const int* dst = eidx + N_EDGES;  // edge_index[1]

    const size_t NODE_ELEMS = (size_t)N_NODES * DIM;   // 6.4M floats
    float* buf0 = (float*)d_ws;
    float* buf1 = buf0 + NODE_ELEMS;
    float* G    = buf1 + NODE_ELEMS;                   // 128 x 128
    float* G1   = G + NUM_GRAPHS * DIM;                // 128 x 128
    float* out  = (float*)d_out;

    const int copyBlocks    = (int)((NODE_ELEMS / 4 + 255) / 256);    // float4 copy
    const int scatterBlocks = (N_EDGES * 32 + 255) / 256;
    const int gemmBlocks    = (N_NODES / 16 + 7) / 8;                 // 8 strips/block
    const int poolBlocks    = (N_NODES * 32 + 255) / 256;

    // ---- GIN layer 1: agg = x + sum_{j->i} x_j ----
    k_copy_f4<<<copyBlocks, 256, 0, stream>>>((const float4*)x, (float4*)buf0,
                                              (int)(NODE_ELEMS / 4));
    k_scatter_add<<<scatterBlocks, 256, 0, stream>>>(x, src, dst, buf0, N_EDGES);
    // h = relu(bn1(agg @ w1_1 + b1_1))
    k_gemm128<<<gemmBlocks, 256, 0, stream>>>(buf0, w1_1, b1_1, bn1_g, bn1_b,
                                              bn1_m, bn1_v, buf1, N_NODES, 0);
    // h = relu(h @ w1_2 + b1_2)
    k_gemm128<<<gemmBlocks, 256, 0, stream>>>(buf1, w1_2, b1_2, b1_2, b1_2,
                                              b1_2, b1_2, buf0, N_NODES, 1);

    // ---- GIN layer 2 ----
    k_copy_f4<<<copyBlocks, 256, 0, stream>>>((const float4*)buf0, (float4*)buf1,
                                              (int)(NODE_ELEMS / 4));
    k_scatter_add<<<scatterBlocks, 256, 0, stream>>>(buf0, src, dst, buf1, N_EDGES);
    k_gemm128<<<gemmBlocks, 256, 0, stream>>>(buf1, w2_1, b2_1, bn2_g, bn2_b,
                                              bn2_m, bn2_v, buf0, N_NODES, 0);
    k_gemm128<<<gemmBlocks, 256, 0, stream>>>(buf0, w2_2, b2_2, b2_2, b2_2,
                                              b2_2, b2_2, buf1, N_NODES, 1);

    // ---- global_add_pool ----
    k_zero_f<<<(NUM_GRAPHS * DIM + 255) / 256, 256, 0, stream>>>(G, NUM_GRAPHS * DIM);
    k_pool_add<<<poolBlocks, 256, 0, stream>>>(buf1, batch, G, N_NODES);

    // ---- head: relu(G @ lin1_w + lin1_b); out = G1 @ lin2_w + lin2_b ----
    k_gemm128<<<1, 256, 0, stream>>>(G, lin1_w, lin1_b, lin1_b, lin1_b,
                                     lin1_b, lin1_b, G1, NUM_GRAPHS, 1);
    k_final_lin<<<1, 256, 0, stream>>>(G1, lin2_w, lin2_b, out);
}
